// LRU_89928025244453
// MI455X (gfx1250) — matrix operand order repaired and checked by reading, not compiled, not run on hardware
//
#include <hip/hip_runtime.h>

// -----------------------------------------------------------------------------
// LRU forward for MI455X (gfx1250, wave32).
//   B=16, L=8192, H=256, N=128.  Bandwidth-bound: compute is ~51 GFLOP even
//   with GEMM recompute, vs 23.3 TB/s HBM.  Traffic-minimal plan (~388 MiB):
//   K0: one-shot fp32->bf16 conversion of Wi/Wo into workspace (256 KiB,
//       L2-resident) -- no weight-conversion VALU in hot loops.
//   K1: per-(b,chunk) input GEMM (v_wmma_f32_16x16x32_bf16) -> LDS -> local
//       complex scan (zero init) -> write ONLY 2 MiB chunk-final states.
//   K2: tiny sequential carry chain across chunks (2048 threads).
//   K3: RECOMPUTE input GEMM (cheap; avoids a 256 MiB locals round-trip),
//       scan seeded with the chunk carry -> true x_t, scale by gamma in-place
//       in LDS, then output GEMM (WMMA) -> d_out (+bo).
// Workspace: finals 2 MiB + carries 2 MiB + bf16 weights 256 KiB.
// -----------------------------------------------------------------------------

#define DEVI __device__ __forceinline__

typedef __attribute__((ext_vector_type(16))) __bf16 v16bf;
typedef __attribute__((ext_vector_type(8)))  float  v8f;

constexpr int Bb     = 16;
constexpr int Ll     = 8192;
constexpr int Hh     = 256;
constexpr int Nn     = 128;
constexpr int TWO_N  = 256;
constexpr int CHUNK  = 64;           // timesteps per workgroup (64 KiB LDS tile)
constexpr int NCHUNK = Ll / CHUNK;   // 128 chunks per batch row

// A fragment (16x32 bf16, ISA 7.12.2): lane holds row M=lane&15;
// element e maps to K = k0 + (e<8?0:16) + 8*half + (e&7)  (two runs of 8
// consecutive fp32 -> two b128 loads + packed v_cvt_pk_bf16_f32).
DEVI v16bf a_frag(const float* rowbase, int half, int k0) {
    v16bf a;
#pragma unroll
    for (int e = 0; e < 16; ++e) {
        int k = k0 + ((e < 8) ? 0 : 16) + 8 * half + (e & 7);
        a[e] = (__bf16)rowbase[k];
    }
    return a;
}

// B fragment (32x16 bf16): lane holds column N=lane&15; element e maps to
// K = k0 + (e<8?0:16) + 8*half + (e&7).  Pre-converted bf16 weights: two runs
// of 8 elements = two aligned b128 loads, no conversion.
DEVI v16bf b_frag(const __bf16* wrow, int half, int k0) {
    v16bf b;
    const int kb = k0 + 8 * half;
#pragma unroll
    for (int e = 0; e < 16; ++e) b[e] = wrow[kb + ((e < 8) ? 0 : 16) + (e & 7)];
    return b;
}

// Shared GEMM tile: dst[0:64][0:256] = A(64x256 fp32) * W^T (bf16) + bias.
// 8 waves: wave = {mt (4 M-tiles of 16 rows)} x {nh (2 halves of 16 N-tiles)}.
DEVI void gemm_tile(const float* __restrict__ abase, const __bf16* __restrict__ WB,
                    const float* __restrict__ bias, float (*dst)[TWO_N], int tid) {
    const int wave = tid >> 5, lane = tid & 31;
    const int half = lane >> 4, lrow = lane & 15;
    const int mt   = wave & 3;
    const int nh   = wave >> 2;
    const float* arow = abase + (size_t)(mt * 16 + lrow) * Hh;
    __builtin_prefetch(arow, 0, 0);   // global_prefetch_b8

    for (int ng = 0; ng < 2; ++ng) {
        v8f acc[4];
#pragma unroll
        for (int j = 0; j < 4; ++j) acc[j] = {};
#pragma unroll 2
        for (int k0 = 0; k0 < Hh; k0 += 32) {
            v16bf a = a_frag(arow, half, k0);
#pragma unroll
            for (int j = 0; j < 4; ++j) {
                int ncol = (nh * 8 + ng * 4 + j) * 16 + lrow;
                v16bf bf = b_frag(WB + (size_t)ncol * Hh, half, k0);
                acc[j] = __builtin_amdgcn_wmma_f32_16x16x32_bf16(
                    false, a, false, bf, (short)0, acc[j], false, false);
            }
        }
        // C/D layout: lane -> column lane&15; rows = vgpr (+8 for upper half).
#pragma unroll
        for (int j = 0; j < 4; ++j) {
            int   ncol = (nh * 8 + ng * 4 + j) * 16 + lrow;
            float bv   = bias[ncol];
#pragma unroll
            for (int v = 0; v < 8; ++v) {
                int mrow = mt * 16 + v + half * 8;
                dst[mrow][ncol] = acc[j][v] + bv;
            }
        }
    }
}

// lambda = exp(-exp(nu_log) + i*exp(theta_log)) for channel n.
DEVI void lambda_of(const float* plog, int n, float& lr, float& li) {
    float nu  = __expf(plog[n]);
    float th  = __expf(plog[Nn + n]);
    float mag = __expf(-nu);
    lr = mag * __cosf(th);
    li = mag * __sinf(th);
}

// -----------------------------------------------------------------------------
// Kernel 0: convert Wi (2N x H) and Wo (H x 2N) to bf16 in workspace.
// -----------------------------------------------------------------------------
__global__ __launch_bounds__(256)
void lru_k0(const float* __restrict__ Wi, const float* __restrict__ Wo,
            __bf16* __restrict__ WiB, __bf16* __restrict__ WoB) {
    int g = blockIdx.x * blockDim.x + threadIdx.x;
    constexpr int NW = TWO_N * Hh;           // 65536 per matrix
    if (g < NW) {
        WiB[g] = (__bf16)Wi[g];
        WoB[g] = (__bf16)Wo[g];
    }
}

// -----------------------------------------------------------------------------
// Kernel 1: u = inputs @ Wi^T + bi (WMMA), local scan with zero init,
//           write only the chunk-final state.
// -----------------------------------------------------------------------------
__global__ __launch_bounds__(256)
void lru_k1(const float* __restrict__ inputs, const __bf16* __restrict__ WiB,
            const float* __restrict__ bi, const float* __restrict__ plog,
            float2* __restrict__ finals) {
    __shared__ float u_lds[CHUNK][TWO_N];   // 64 KiB

    const int blk = blockIdx.x;
    const int b   = blk / NCHUNK;
    const int c   = blk % NCHUNK;
    const int t0  = c * CHUNK;
    const int tid = threadIdx.x;

    gemm_tile(inputs + ((size_t)b * Ll + t0) * Hh, WiB, bi, u_lds, tid);
    __syncthreads();

    if (tid < Nn) {
        const int n = tid;
        float lr, li; lambda_of(plog, n, lr, li);
        float sr = 0.f, si = 0.f;
#pragma unroll 4
        for (int t = 0; t < CHUNK; ++t) {
            float ur = u_lds[t][n];
            float ui = u_lds[t][n + Nn];
            float nr = fmaf(lr, sr, fmaf(-li, si, ur));
            float ni = fmaf(lr, si, fmaf( li, sr, ui));
            sr = nr; si = ni;
        }
        finals[((size_t)b * NCHUNK + c) * Nn + n] = make_float2(sr, si);
    }
}

// -----------------------------------------------------------------------------
// Kernel 2: carry chain across chunks: carry_c = lambda^CHUNK * carry_{c-1}
//           + final_{c-1}. One thread per (b, n).
// -----------------------------------------------------------------------------
__global__ __launch_bounds__(256)
void lru_k2(const float* __restrict__ plog, const float2* __restrict__ finals,
            float2* __restrict__ carries) {
    int g = blockIdx.x * blockDim.x + threadIdx.x;
    if (g >= Bb * Nn) return;
    int b = g / Nn, n = g % Nn;
    float nu  = __expf(plog[n]);
    float th  = __expf(plog[Nn + n]);
    float mag = __expf(-(float)CHUNK * nu);
    float ang = (float)CHUNK * th;
    float Lr  = mag * __cosf(ang);
    float Li  = mag * __sinf(ang);
    float cr = 0.f, ci = 0.f;
    const float2* fin = finals  + (size_t)b * NCHUNK * Nn + n;
    float2*       car = carries + (size_t)b * NCHUNK * Nn + n;
    for (int c = 0; c < NCHUNK; ++c) {
        car[(size_t)c * Nn] = make_float2(cr, ci);
        float2 f  = fin[(size_t)c * Nn];
        float  nr = fmaf(Lr, cr, fmaf(-Li, ci, f.x));
        float  ni = fmaf(Lr, ci, fmaf( Li, cr, f.y));
        cr = nr; ci = ni;
    }
}

// -----------------------------------------------------------------------------
// Kernel 3: recompute u (WMMA), scan seeded with carry -> x_t, scale by gamma
//           in-place in LDS, output GEMM (WMMA) -> d_out (+bo).
// -----------------------------------------------------------------------------
__global__ __launch_bounds__(256)
void lru_k3(const float* __restrict__ inputs, const __bf16* __restrict__ WiB,
            const float* __restrict__ bi, const __bf16* __restrict__ WoB,
            const float* __restrict__ bo, const float* __restrict__ plog,
            const float2* __restrict__ carries, float* __restrict__ out) {
    __shared__ float tile[CHUNK][TWO_N];   // 64 KiB: holds u, then y in place

    const int blk = blockIdx.x;
    const int b   = blk / NCHUNK;
    const int c   = blk % NCHUNK;
    const int t0  = c * CHUNK;
    const int tid = threadIdx.x;

    // Phase 1: u = inputs @ Wi^T + bi (recompute; avoids 256 MiB round-trip).
    gemm_tile(inputs + ((size_t)b * Ll + t0) * Hh, WiB, bi, tile, tid);
    __syncthreads();

    // Phase 2: scan seeded with carry; overwrite in place with y = gamma*x.
    // Thread n exclusively owns columns n and n+128 -> in-place is race-free.
    if (tid < Nn) {
        const int n = tid;
        float lr, li; lambda_of(plog, n, lr, li);
        float gam = __expf(plog[2 * Nn + n]);
        float2 cv = carries[((size_t)b * NCHUNK + c) * Nn + n];
        float sr = cv.x, si = cv.y;
#pragma unroll 4
        for (int t = 0; t < CHUNK; ++t) {
            float ur = tile[t][n];
            float ui = tile[t][n + Nn];
            float nr = fmaf(lr, sr, fmaf(-li, si, ur));
            float ni = fmaf(lr, si, fmaf( li, sr, ui));
            sr = nr; si = ni;
            tile[t][n]      = gam * sr;
            tile[t][n + Nn] = gam * si;
        }
    }
    __syncthreads();

    // Phase 3: out = y @ Wo^T + bo.
    const int wave = tid >> 5, lane = tid & 31;
    const int half = lane >> 4, lrow = lane & 15;
    const int mt   = wave & 3;
    const int nh   = wave >> 2;
    const float* arow = &tile[mt * 16 + lrow][0];

    for (int ng = 0; ng < 2; ++ng) {
        v8f acc[4];
#pragma unroll
        for (int j = 0; j < 4; ++j) acc[j] = {};
#pragma unroll 2
        for (int k0 = 0; k0 < TWO_N; k0 += 32) {
            v16bf a = a_frag(arow, half, k0);   // ds_load path (LDS source)
#pragma unroll
            for (int j = 0; j < 4; ++j) {
                int ncol = (nh * 8 + ng * 4 + j) * 16 + lrow;   // output h
                v16bf bf = b_frag(WoB + (size_t)ncol * TWO_N, half, k0);
                acc[j] = __builtin_amdgcn_wmma_f32_16x16x32_bf16(
                    false, a, false, bf, (short)0, acc[j], false, false);
            }
        }
#pragma unroll
        for (int j = 0; j < 4; ++j) {
            int   ncol = (nh * 8 + ng * 4 + j) * 16 + lrow;
            float bias = bo[ncol];
#pragma unroll
            for (int v = 0; v < 8; ++v) {
                int mglob = t0 + mt * 16 + v + half * 8;
                out[((size_t)b * Ll + mglob) * Hh + ncol] = acc[j][v] + bias;
            }
        }
    }
}

// -----------------------------------------------------------------------------
extern "C" void kernel_launch(void* const* d_in, const int* in_sizes, int n_in,
                              void* d_out, int out_size, void* d_ws, size_t ws_size,
                              hipStream_t stream) {
    (void)in_sizes; (void)n_in; (void)out_size; (void)ws_size;
    const float* inputs = (const float*)d_in[0];
    const float* Wi     = (const float*)d_in[1];
    const float* bi     = (const float*)d_in[2];
    const float* Wo     = (const float*)d_in[3];
    const float* bo     = (const float*)d_in[4];
    const float* plog   = (const float*)d_in[5];
    float*       out    = (float*)d_out;

    // Workspace layout (~4.3 MiB): finals | carries | WiB | WoB
    float2* finals  = (float2*)d_ws;
    float2* carries = finals + (size_t)Bb * NCHUNK * Nn;
    __bf16* WiB     = (__bf16*)(carries + (size_t)Bb * NCHUNK * Nn);
    __bf16* WoB     = WiB + (size_t)TWO_N * Hh;

    const int blocks = Bb * NCHUNK;   // 2048 workgroups of 256 threads (8 waves)
    lru_k0<<<(TWO_N * Hh + 255) / 256, 256, 0, stream>>>(Wi, Wo, WiB, WoB);
    lru_k1<<<blocks, 256, 0, stream>>>(inputs, WiB, bi, plog, finals);
    lru_k2<<<(Bb * Nn + 255) / 256, 256, 0, stream>>>(plog, finals, carries);
    lru_k3<<<blocks, 256, 0, stream>>>(inputs, WiB, bi, WoB, bo, plog, carries, out);
}
